// SDVAE_14955076125396
// MI455X (gfx1250) — matrix operand order repaired and checked by reading, not compiled, not run on hardware
//
#include <hip/hip_runtime.h>
#include <hip/hip_bf16.h>

// ---------------------------------------------------------------------------
// RelationalDynamics forward for MI455X (gfx1250, wave32, WMMA).
//
// Roofline: ~35.5 GFLOP, ~50 MB HBM traffic -> dominant pairwise GEMM
// (B*28 x 256 x 256) runs on v_wmma_f32_16x16x32_f16 with weights resident
// in LDS (f32 accumulate, f16 operands). All LDS staging is vectorized
// (float4 loads -> packed f16 stores); next-tile global reads are prefetched
// with global_prefetch_b8 while WMMAs run.
//
// Workspace layout (d_ws, needs 50,331,648 B):
//   zE : _Float16 [65536][128]  (16.8 MB)  = ELU(z @ Wobj^T + bobj)
//   Eg : _Float16 [65536][256]  (33.5 MB)  = per-object summed weighted effects
// ---------------------------------------------------------------------------

typedef __attribute__((ext_vector_type(16))) _Float16 v16h;
typedef __attribute__((ext_vector_type(8)))  _Float16 v8h;
typedef __attribute__((ext_vector_type(4)))  _Float16 v4h;
typedef __attribute__((ext_vector_type(8)))  float    v8f;
typedef __attribute__((ext_vector_type(4)))  float    v4f;

__device__ __forceinline__ v8f wmma_f16(v16h a, v16h b, v8f c) {
  // (neg_a, A, neg_b, B, c_mod, C, reuse_a, reuse_b)
  return __builtin_amdgcn_wmma_f32_16x16x32_f16(false, a, false, b,
                                                (short)0, c, false, false);
}

// A fragment (16x32 f16) from LDS, row-major [row][k], ld in f16 units.
// ISA layout: lanes 0-15 hold row M=lane, K = k0+{0..7, 16..23};
//             lanes 16-31 hold row M=lane-16, K = k0+{8..15, 24..31}.
__device__ __forceinline__ v16h load_a_frag(const _Float16* base, int row0,
                                            int k0, int ld) {
  const int lane = threadIdx.x & 31;
  const _Float16* p = base + (row0 + (lane & 15)) * ld + k0 + ((lane >> 4) << 3);
  v8h lo = *(const v8h*)(p);        // K offsets +0..+7
  v8h hi = *(const v8h*)(p + 16);   // K offsets +16..+23
  v16h a;
#pragma unroll
  for (int t = 0; t < 8; ++t) { a[t] = lo[t]; a[t + 8] = hi[t]; }
  return a;
}

// B fragment (32x16 f16, K x N) from LDS where W is stored [N][K] row-major
// (i.e. the PyTorch Linear weight layout, converted to f16).
// Lanes 0-15: column N=lane, K = k0+{0..7, 16..23}; lanes 16-31: same N, K = k0+{8..15, 24..31}.
__device__ __forceinline__ v16h load_b_frag(const _Float16* base, int k0,
                                            int col0, int ld) {
  const int lane = threadIdx.x & 31;
  const _Float16* p = base + (col0 + (lane & 15)) * ld + k0 + ((lane >> 4) << 3);
  v8h lo = *(const v8h*)(p);
  v8h hi = *(const v8h*)(p + 16);
  v16h b;
#pragma unroll
  for (int t = 0; t < 8; ++t) { b[t] = lo[t]; b[t + 8] = hi[t]; }
  return b;
}

__device__ __forceinline__ v4h cvt4(v4f v) {
  v4h h;
#pragma unroll
  for (int t = 0; t < 4; ++t) h[t] = (_Float16)v[t];
  return h;
}

__device__ __forceinline__ float elu_f(float x) {
  return x > 0.f ? x : (__expf(x) - 1.f);
}

// ===========================================================================
// Kernel 1: z_embed = ELU(z @ Wobj^T + bobj)   [65536,64] -> f16 [65536,128]
// Block: 256 threads (8 waves), 128 rows per block. Grid 512.
// ===========================================================================
__global__ void k_embed(const float* __restrict__ z,
                        const float* __restrict__ Wobj,
                        const float* __restrict__ bobj,
                        _Float16* __restrict__ zE) {
  __shared__ _Float16 Az[128 * 72];   // z tile, f16, ld 72
  __shared__ _Float16 Wl[128 * 72];   // Wobj [128 out][64 in], f16, ld 72
  __shared__ float    bias[128];
  const int tid = threadIdx.x;
  const int m0  = blockIdx.x * 128;

  // 128 rows x 16 float4 per row (z and Wobj both 128x64 f32)
  for (int idx = tid; idx < 128 * 16; idx += 256) {
    int r = idx >> 4, c4 = (idx & 15) << 2;
    v4f vz = *(const v4f*)&z[(size_t)(m0 + r) * 64 + c4];
    v4f vw = *(const v4f*)&Wobj[r * 64 + c4];
    *(v4h*)&Az[r * 72 + c4] = cvt4(vz);
    *(v4h*)&Wl[r * 72 + c4] = cvt4(vw);
  }
  if (tid < 128) bias[tid] = bobj[tid];
  __syncthreads();

  const int wv = tid >> 5, lane = tid & 31;
  v16h a0 = load_a_frag(Az, 16 * wv, 0, 72);
  v16h a1 = load_a_frag(Az, 16 * wv, 32, 72);
#pragma unroll
  for (int nt = 0; nt < 8; ++nt) {
    v16h b0 = load_b_frag(Wl, 0, 16 * nt, 72);
    v16h b1 = load_b_frag(Wl, 32, 16 * nt, 72);
    v8f c = {};
    c = wmma_f16(a0, b0, c);
    c = wmma_f16(a1, b1, c);
    const int col = nt * 16 + (lane & 15);
    const int rb  = (lane >> 4) * 8;
#pragma unroll
    for (int r = 0; r < 8; ++r) {
      float v = elu_f(c[r] + bias[col]);
      zE[(size_t)(m0 + 16 * wv + rb + r) * 128 + col] = (_Float16)v;
    }
  }
}

// ===========================================================================
// Kernel 2: per batch b — eff = ELU(zed @ We^T + be); att = sigmoid(eff.Wa+ba)
//           E[k] = sum_{pairs containing k} att*eff;  write Eg f16 [B*8][256].
// We (128 KB f16) cached in LDS; each block loops over 8 batches, prefetching
// the next batch's embeddings during the GEMM.
// Grid 1024, block 256 (8 waves). Dynamic LDS: 190,976 B.
// ===========================================================================
#define LD2 264  // padded ld (f16/f32 elements) for 256-wide arrays

__global__ void k_pairs(const _Float16* __restrict__ zE,
                        const float* __restrict__ We,
                        const float* __restrict__ be,
                        const float* __restrict__ Wa,
                        const float* __restrict__ ba,
                        _Float16* __restrict__ Eg,
                        int bpb) {
  extern __shared__ __align__(16) char smem[];
  _Float16* WeL  = (_Float16*)(smem);                    // 256*264*2 = 135168
  _Float16* Az   = (_Float16*)(smem + 135168);           // 32*264*2  =  16896
  float*    effL = (float*)   (smem + 152064);           // 32*264*4  =  33792
  float*    beL  = (float*)   (smem + 185856);           // 1024
  float*    WaL  = (float*)   (smem + 186880);           // 1024
  float*    part = (float*)   (smem + 187904);           // 28*8*4 = 896
  float*    attL = (float*)   (smem + 188800);           // 128
  _Float16* zloc = (_Float16*)(smem + 188928);           // 8*128*2 = 2048

  constexpr int IU[28] = {0,0,0,0,0,0,0, 1,1,1,1,1,1, 2,2,2,2,2,
                          3,3,3,3, 4,4,4, 5,5, 6};
  constexpr int JU[28] = {1,2,3,4,5,6,7, 2,3,4,5,6,7, 3,4,5,6,7,
                          4,5,6,7, 5,6,7, 6,7, 7};

  const int tid = threadIdx.x;
  const int wv = tid >> 5, lane = tid & 31;

  // Load We [256 out][256 in] -> f16 LDS (reused for all batches of block)
  for (int idx = tid; idx < 256 * 64; idx += 256) {
    int r = idx >> 6, c4 = (idx & 63) << 2;
    *(v4h*)&WeL[r * LD2 + c4] = cvt4(*(const v4f*)&We[(r << 8) + c4]);
  }
  if (tid < 256) { beL[tid] = be[tid]; WaL[tid] = Wa[tid]; }

  for (int it = 0; it < bpb; ++it) {
    const int b = blockIdx.x * bpb + it;
    __syncthreads();  // weights ready / previous iteration's LDS consumers done

    const _Float16* zp = zE + (size_t)b * 8 * 128;
    for (int idx = tid; idx < 128; idx += 256)   // 128 chunks of 8 f16
      *(v8h*)&zloc[idx << 3] = *(const v8h*)&zp[idx << 3];
    __syncthreads();

    // Build zed: rows = 28 pairs (padded to 32), cols = [z_i | z_j]
    for (int idx = tid; idx < 32 * 32; idx += 256) {
      int r = idx >> 5, c8 = (idx & 31) << 3;
      v8h v = {};
      if (r < 28)
        v = (c8 < 128) ? *(const v8h*)&zloc[IU[r] * 128 + c8]
                       : *(const v8h*)&zloc[JU[r] * 128 + (c8 - 128)];
      *(v8h*)&Az[r * LD2 + c8] = v;
    }
    // Prefetch next batch's embeddings (global_prefetch_b8) while GEMM runs.
    if (it + 1 < bpb)
      __builtin_prefetch(zp + 8 * 128 + (tid << 2), 0, 0);
    __syncthreads();

    // GEMM 32x256x256 (f16 WMMA, f32 acc) + bias + ELU -> effL (f32)
#pragma unroll
    for (int mt = 0; mt < 2; ++mt) {
      v16h a[8];
#pragma unroll
      for (int ks = 0; ks < 8; ++ks) a[ks] = load_a_frag(Az, 16 * mt, 32 * ks, LD2);
#pragma unroll
      for (int u = 0; u < 2; ++u) {
        const int nt = wv * 2 + u;
        v8f c = {};
#pragma unroll
        for (int ks = 0; ks < 8; ++ks) {
          v16h bf = load_b_frag(WeL, 32 * ks, 16 * nt, LD2);
          c = wmma_f16(a[ks], bf, c);
        }
        const int col = nt * 16 + (lane & 15);
        const int rb  = 16 * mt + (lane >> 4) * 8;
#pragma unroll
        for (int r = 0; r < 8; ++r)
          effL[(rb + r) * LD2 + col] = elu_f(c[r] + beL[col]);
      }
    }
    __syncthreads();

    // Attention: att_p = sigmoid(dot(eff_p, Wa) + ba)
    if (tid < 224) {
      const int p = tid >> 3, seg = tid & 7;
      float s = 0.f;
#pragma unroll
      for (int t = 0; t < 32; ++t) {
        int n = seg * 32 + t;
        s += effL[p * LD2 + n] * WaL[n];
      }
      part[p * 8 + seg] = s;
    }
    __syncthreads();
    if (tid < 28) {
      float s = ba[0];
#pragma unroll
      for (int t = 0; t < 8; ++t) s += part[tid * 8 + t];
      attL[tid] = 1.f / (1.f + __expf(-s));
    }
    __syncthreads();

    // E[k][n] = sum over pairs containing k of att*eff  (one column per thread)
    {
      const int n = tid;
      float E[8] = {0.f, 0.f, 0.f, 0.f, 0.f, 0.f, 0.f, 0.f};
#pragma unroll
      for (int p = 0; p < 28; ++p) {
        float w = attL[p] * effL[p * LD2 + n];
        E[IU[p]] += w;
        E[JU[p]] += w;
      }
      _Float16* ep = Eg + (size_t)b * 8 * 256 + n;
#pragma unroll
      for (int k = 0; k < 8; ++k) ep[k * 256] = (_Float16)E[k];
    }
  }
}

// ===========================================================================
// Kernel 3: rel = concat(zE, Eg) @ Wc^T + bc; out = [rel@Wloc^T+bloc |
//           rel@Wscale^T+bscale] (f32). 64 rows per chunk, weights in LDS,
// next chunk's activations prefetched during the GEMMs.
// Grid 512, block 256 (8 waves), chunks 1024. Dynamic LDS: 128,768 B.
// ===========================================================================
__global__ void k_comb(const _Float16* __restrict__ zE,
                       const _Float16* __restrict__ Eg,
                       const float* __restrict__ Wc,  const float* __restrict__ bc,
                       const float* __restrict__ Wloc, const float* __restrict__ bloc,
                       const float* __restrict__ Wscale, const float* __restrict__ bscale,
                       float* __restrict__ out, int nchunks) {
  extern __shared__ __align__(16) char smem[];
  _Float16* WcL  = (_Float16*)(smem);                    // 64*392*2 = 50176
  _Float16* A    = (_Float16*)(smem + 50176);            // 64*392*2 = 50176
  _Float16* WlL  = (_Float16*)(smem + 100352);           // 64*72*2 = 9216
  _Float16* WsL  = (_Float16*)(smem + 109568);           // 9216
  _Float16* relL = (_Float16*)(smem + 118784);           // 9216
  float*    bcL  = (float*)   (smem + 128000);           // 256
  float*    blL  = (float*)   (smem + 128256);           // 256
  float*    bsL  = (float*)   (smem + 128512);           // 256

  const int tid = threadIdx.x;
  const int wv = tid >> 5, lane = tid & 31;

  // Wc: 64 rows x 96 float4; Wloc/Wscale: 64 rows x 16 float4
  for (int idx = tid; idx < 64 * 96; idx += 256) {
    int r = idx / 96, c4 = (idx - r * 96) << 2;
    *(v4h*)&WcL[r * 392 + c4] = cvt4(*(const v4f*)&Wc[r * 384 + c4]);
  }
  for (int idx = tid; idx < 64 * 16; idx += 256) {
    int r = idx >> 4, c4 = (idx & 15) << 2;
    *(v4h*)&WlL[r * 72 + c4] = cvt4(*(const v4f*)&Wloc[(r << 6) + c4]);
    *(v4h*)&WsL[r * 72 + c4] = cvt4(*(const v4f*)&Wscale[(r << 6) + c4]);
  }
  if (tid < 64) { bcL[tid] = bc[tid]; blL[tid] = bloc[tid]; bsL[tid] = bscale[tid]; }

  for (int ch = blockIdx.x; ch < nchunks; ch += gridDim.x) {
    const int r0 = ch * 64;
    __syncthreads();  // weights ready / previous chunk done

    // A = [ zE (128) | Eg (256) ]  f16, 64 rows x 48 chunks of 8
    for (int idx = tid; idx < 64 * 48; idx += 256) {
      int r = idx / 48, c8 = (idx - r * 48) << 3;
      v8h v = (c8 < 128) ? *(const v8h*)&zE[(size_t)(r0 + r) * 128 + c8]
                         : *(const v8h*)&Eg[(size_t)(r0 + r) * 256 + (c8 - 128)];
      *(v8h*)&A[r * 392 + c8] = v;
    }
    // Prefetch next chunk's activations while GEMMs run.
    {
      const int chn = ch + gridDim.x;
      if (chn < nchunks) {
        const int rn = chn * 64;
        __builtin_prefetch(zE + (size_t)rn * 128 + (tid << 5), 0, 0); // 16 KB
        __builtin_prefetch(Eg + (size_t)rn * 256 + (tid << 6), 0, 0); // 32 KB
      }
    }
    __syncthreads();

    // GEMM1: 64x384 @ 384x64 -> rel (f16 in LDS). 16 tiles, 2 per wave.
#pragma unroll
    for (int u = 0; u < 2; ++u) {
      const int id = wv * 2 + u;
      const int mt = id >> 2, nt = id & 3;
      v8f c = {};
#pragma unroll
      for (int ks = 0; ks < 12; ++ks) {
        v16h af = load_a_frag(A, 16 * mt, 32 * ks, 392);
        v16h bf = load_b_frag(WcL, 32 * ks, 16 * nt, 392);
        c = wmma_f16(af, bf, c);
      }
      const int col = nt * 16 + (lane & 15);
      const int rb  = 16 * mt + (lane >> 4) * 8;
#pragma unroll
      for (int r = 0; r < 8; ++r)
        relL[(rb + r) * 72 + col] = (_Float16)(c[r] + bcL[col]);
    }
    __syncthreads();

    // GEMM2: rel @ {Wloc,Wscale}^T -> out halves. 32 tiles, 4 per wave.
#pragma unroll
    for (int u = 0; u < 4; ++u) {
      const int id = wv * 4 + u;
      const int sel = id >> 4;            // 0 = loc, 1 = scale
      const int mt = (id >> 2) & 3, nt = id & 3;
      const _Float16* W = sel ? WsL : WlL;
      const float*    bb = sel ? bsL : blL;
      v8f c = {};
#pragma unroll
      for (int ks = 0; ks < 2; ++ks) {
        v16h af = load_a_frag(relL, 16 * mt, 32 * ks, 72);
        v16h bf = load_b_frag(W, 32 * ks, 16 * nt, 72);
        c = wmma_f16(af, bf, c);
      }
      const int col = nt * 16 + (lane & 15);
      const int rb  = 16 * mt + (lane >> 4) * 8;
#pragma unroll
      for (int r = 0; r < 8; ++r)
        out[(size_t)(r0 + rb + r) * 128 + sel * 64 + col] = c[r] + bb[col];
    }
    __syncthreads();
  }
}

// ===========================================================================
extern "C" void kernel_launch(void* const* d_in, const int* in_sizes, int n_in,
                              void* d_out, int out_size, void* d_ws, size_t ws_size,
                              hipStream_t stream) {
  const float* z      = (const float*)d_in[0];
  // d_in[1] = h (unused by the reference computation)
  const float* Wobj   = (const float*)d_in[2];
  const float* bobj   = (const float*)d_in[3];
  const float* We     = (const float*)d_in[4];
  const float* be     = (const float*)d_in[5];
  const float* Wa     = (const float*)d_in[6];
  const float* ba     = (const float*)d_in[7];
  const float* Wc     = (const float*)d_in[8];
  const float* bc     = (const float*)d_in[9];
  const float* Wloc   = (const float*)d_in[10];
  const float* bloc   = (const float*)d_in[11];
  const float* Wscale = (const float*)d_in[12];
  const float* bscale = (const float*)d_in[13];
  float* out = (float*)d_out;

  _Float16* zE = (_Float16*)d_ws;                                    // 16.8 MB
  _Float16* Eg = (_Float16*)((char*)d_ws + (size_t)65536 * 128 * 2); // 33.5 MB

  k_embed<<<512, 256, 0, stream>>>(z, Wobj, bobj, zE);

  const size_t sm2 = 190976;  // dynamic LDS (WGP has 320 KB)
  k_pairs<<<1024, 256, sm2, stream>>>(zE, We, be, Wa, ba, Eg, 8);

  const size_t sm3 = 128768;
  k_comb<<<512, 256, sm3, stream>>>(zE, Eg, Wc, bc, Wloc, bloc, Wscale, bscale,
                                    out, 1024);
}
